// MCRec_84018150244392
// MI455X (gfx1250) — compile-verified, matrix operand order checked by reading, not run on hardware
//
#include <hip/hip_runtime.h>
#include <hip/hip_bf16.h>
#include <math.h>

// ---------------- problem constants ----------------
constexpr int MP   = 3;      // metapaths
constexpr int B    = 8192;   // batch
constexpr int PT   = 20;     // P*T
constexpr int FDIM = 128;    // feature
constexpr int LDIM = 128;    // latent

// ---------------- WMMA types ----------------
typedef __bf16 bf16_t;
typedef bf16_t v16bf __attribute__((ext_vector_type(16)));
typedef float  v8f   __attribute__((ext_vector_type(8)));

static __device__ inline v8f zero8() {
    v8f c;
#pragma unroll
    for (int i = 0; i < 8; ++i) c[i] = 0.0f;
    return c;
}

// A fragment: 16x32 bf16, lane (r = lane&15, h = lane>>4) holds row r,
// K = {k0 + 8h + e} (elems 0..7) and {k0 + 16 + 8h + e} (elems 8..15).
// B fragment identical with "row" = output column, reading from sB[n][k] (transposed-B layout).
static __device__ inline v16bf load_frag(const bf16_t* base, int ld, int rc0, int k0, int lane) {
    const int r   = lane & 15;
    const int hlf = lane >> 4;
    const bf16_t* p = base + (size_t)(rc0 + r) * ld + k0 + hlf * 8;
    v16bf f;
#pragma unroll
    for (int e = 0; e < 8; ++e) { f[e] = p[e]; f[8 + e] = p[e + 16]; }
    return f;
}

// ---- CDNA5 async global->LDS copy (ASYNCcnt path), GVS addressing ----
// LDS destination address in lds_off ; mem = SGPR base + 32-bit per-lane byte offset
static __device__ inline void async_load_b128(unsigned lds_off, const void* base, int byte_off) {
    asm volatile("global_load_async_to_lds_b128 %0, %1, %2"
                 :
                 : "v"(lds_off), "v"(byte_off), "s"((unsigned long long)(uintptr_t)base)
                 : "memory");
}
static __device__ inline void wait_async0() {
    asm volatile("s_wait_asynccnt 0" ::: "memory");
}

// ================= kernel 1: gather user/item embeddings =================
__global__ void k_gather(const int* __restrict__ ui, const int* __restrict__ ii,
                         const float* __restrict__ uemb, const float* __restrict__ iemb,
                         float* __restrict__ ul, float* __restrict__ il) {
    int b = blockIdx.x, t = threadIdx.x;
    ul[(size_t)b * LDIM + t] = uemb[(size_t)ui[b] * LDIM + t];
    il[(size_t)b * LDIM + t] = iemb[(size_t)ii[b] * LDIM + t];
}

// ================= kernel 2: conv (einsum) + max over (p,t)  =================
// 1 WG = (m, 4 consecutive b) -> A is 80x128 (5 row-tiles), Wconv[m] is B (128x128),
// bf16 WMMA w/ f32 accumulate, then max over each group of 20 rows + bconv.
__global__ __launch_bounds__(256) void k_conv_pool(const float* __restrict__ path,
                                                   const float* __restrict__ Wconv,
                                                   const float* __restrict__ bconv,
                                                   float* __restrict__ pl) {
    constexpr int BB   = 4;
    constexpr int ROWS = BB * PT; // 80
    __shared__ __align__(16) unsigned char smem[ROWS * FDIM * 2 + LDIM * FDIM * 2]; // 53248 B
    bf16_t* sA = (bf16_t*)smem;                            // [80][128]
    bf16_t* sB = (bf16_t*)(smem + ROWS * FDIM * 2);        // [l][f] == Wconv[m] native layout
    float*  sC = (float*)smem;                             // [80][128] (aliased after barrier)

    const int m   = blockIdx.x / (B / BB);
    const int b0  = (blockIdx.x % (B / BB)) * BB;
    const int tid = threadIdx.x;

    // stage A: path[m][b0..b0+3][p][t][f] -> bf16
    const float4* src = (const float4*)(path + ((size_t)m * B + b0) * (PT * FDIM));
    for (int idx = tid; idx < ROWS * FDIM / 4; idx += 256) {
        float4 v = src[idx];
        int base = idx * 4;
        sA[base + 0] = (bf16_t)v.x; sA[base + 1] = (bf16_t)v.y;
        sA[base + 2] = (bf16_t)v.z; sA[base + 3] = (bf16_t)v.w;
    }
    // stage W: Wconv[m][l][f] -> sB[l][f] (this IS the transposed-B layout we want)
    const float4* wsrc = (const float4*)(Wconv + (size_t)m * LDIM * FDIM);
    for (int idx = tid; idx < LDIM * FDIM / 4; idx += 256) {
        float4 v = wsrc[idx];
        int base = idx * 4;
        sB[base + 0] = (bf16_t)v.x; sB[base + 1] = (bf16_t)v.y;
        sB[base + 2] = (bf16_t)v.z; sB[base + 3] = (bf16_t)v.w;
    }
    __syncthreads();

    const int wave = tid >> 5, lane = tid & 31;
    v8f acc[5];
#pragma unroll
    for (int i = 0; i < 5; ++i) {
        int t = wave * 5 + i;           // 40 tiles over 8 waves
        int mt = t >> 3, nt = t & 7;
        v8f c = zero8();
#pragma unroll
        for (int kk = 0; kk < 4; ++kk) {
            v16bf a = load_frag(sA, FDIM, mt * 16, kk * 32, lane);
            v16bf b = load_frag(sB, FDIM, nt * 16, kk * 32, lane);
            c = __builtin_amdgcn_wmma_f32_16x16x32_bf16(false, a, false, b, (short)0, c,
                                                        false, false);
        }
        acc[i] = c;
    }
    __syncthreads();   // all LDS reads done; safe to alias sC over sA/sB

    const int r = lane & 15, hlf = lane >> 4;
#pragma unroll
    for (int i = 0; i < 5; ++i) {
        int t = wave * 5 + i;
        int mt = t >> 3, nt = t & 7;
        int col = nt * 16 + r;
#pragma unroll
        for (int e = 0; e < 8; ++e) sC[(mt * 16 + hlf * 8 + e) * LDIM + col] = acc[i][e];
    }
    __syncthreads();

    // max over the 20 (p,t) rows per b, then + bconv (bias is (p,t)-invariant)
    for (int o = tid; o < BB * LDIM; o += 256) {
        int bl = o >> 7, l = o & 127;
        const float* c = sC + bl * PT * LDIM + l;
        float v = -3.4e38f;
#pragma unroll
        for (int rr = 0; rr < PT; ++rr) v = fmaxf(v, c[rr * LDIM]);
        pl[((size_t)m * B + b0 + bl) * LDIM + l] = v + bconv[m * LDIM + l];
    }
}

// ================= kernel 3: h = relu([ul|il|pl] @ W1 + b1); score = relu(h@W2+b2) ==========
// 1 WG = (m, 64 b rows). K split into 3 chunks of 128 sourced from ul / il / pl.
__global__ __launch_bounds__(256) void k_score(const float* __restrict__ ul,
                                               const float* __restrict__ il,
                                               const float* __restrict__ pl,
                                               const float* __restrict__ W1,
                                               const float* __restrict__ b1,
                                               const float* __restrict__ W2,
                                               const float* __restrict__ b2,
                                               float* __restrict__ score) {
    __shared__ __align__(16) unsigned char smem[64 * 128 * 2 + 128 * 128 * 2]; // 48 KB
    bf16_t* sA = (bf16_t*)smem;                      // [64][128]
    bf16_t* sB = (bf16_t*)(smem + 64 * 128 * 2);     // [n][k] (transposed W1 chunk)
    float*  sC = (float*)smem;                       // [64][128] aliased at the end

    const int m   = blockIdx.x / (B / 64);
    const int b0  = (blockIdx.x % (B / 64)) * 64;
    const int tid = threadIdx.x;
    const int wave = tid >> 5, lane = tid & 31;

    v8f acc[4];
#pragma unroll
    for (int i = 0; i < 4; ++i) acc[i] = zero8();

    const float* srcs[3] = { ul + (size_t)b0 * LDIM,
                             il + (size_t)b0 * LDIM,
                             pl + ((size_t)m * B + b0) * LDIM };

    for (int c = 0; c < 3; ++c) {
        __syncthreads();  // previous chunk's fragment reads finished
        const float4* s = (const float4*)srcs[c];
        for (int idx = tid; idx < 64 * 128 / 4; idx += 256) {
            float4 v = s[idx];
            int base = idx * 4;
            sA[base + 0] = (bf16_t)v.x; sA[base + 1] = (bf16_t)v.y;
            sA[base + 2] = (bf16_t)v.z; sA[base + 3] = (bf16_t)v.w;
        }
        const float* w = W1 + (size_t)c * 128 * 128;   // rows k0..k0+127 of W1
        for (int idx = tid; idx < 128 * 128; idx += 256) {
            int k = idx >> 7, n = idx & 127;
            sB[n * 128 + k] = (bf16_t)w[idx];          // coalesced read, transpose into LDS
        }
        __syncthreads();

        // prefetch next chunk (W1 + A source) into cache while this chunk's WMMAs run
        if (c < 2) {
            const float* nw = W1 + (size_t)(c + 1) * 128 * 128;   // 64 KB
            for (int i = tid; i < 512; i += 256) __builtin_prefetch(nw + i * 32, 0, 1);
            const float* ns = srcs[c + 1];                        // 32 KB
            for (int i = tid; i < 256; i += 256) __builtin_prefetch(ns + i * 32, 0, 1);
        }

#pragma unroll
        for (int i = 0; i < 4; ++i) {
            int t = wave * 4 + i;                      // 32 tiles over 8 waves
            int mt = t >> 3, nt = t & 7;
#pragma unroll
            for (int kk = 0; kk < 4; ++kk) {
                v16bf a = load_frag(sA, 128, mt * 16, kk * 32, lane);
                v16bf b = load_frag(sB, 128, nt * 16, kk * 32, lane);
                acc[i] = __builtin_amdgcn_wmma_f32_16x16x32_bf16(false, a, false, b, (short)0,
                                                                 acc[i], false, false);
            }
        }
    }
    __syncthreads();

    const int r = lane & 15, hlf = lane >> 4;
#pragma unroll
    for (int i = 0; i < 4; ++i) {
        int t = wave * 4 + i;
        int mt = t >> 3, nt = t & 7;
        int col = nt * 16 + r;
#pragma unroll
        for (int e = 0; e < 8; ++e) {
            int row = mt * 16 + hlf * 8 + e;
            sC[row * 128 + col] = fmaxf(acc[i][e] + b1[col], 0.0f);   // + b1, relu
        }
    }
    __syncthreads();

    if (tid < 64) {
        float a = b2[0];
        for (int n = 0; n < 128; ++n) a = fmaf(sC[tid * 128 + n], W2[n], a);
        score[(size_t)m * B + b0 + tid] = fmaxf(a, 0.0f);
    }
}

// ================= kernel 4: softmax over BATCH axis (per m) =================
__global__ __launch_bounds__(256) void k_softmax_b(const float* __restrict__ score,
                                                   float* __restrict__ att) {
    __shared__ float red[256];
    const int m = blockIdx.x, tid = threadIdx.x;
    const float* s = score + (size_t)m * B;

    float mx = -3.4e38f;
    for (int i = tid; i < B; i += 256) mx = fmaxf(mx, s[i]);
    red[tid] = mx; __syncthreads();
    for (int st = 128; st > 0; st >>= 1) {
        if (tid < st) red[tid] = fmaxf(red[tid], red[tid + st]);
        __syncthreads();
    }
    mx = red[0]; __syncthreads();

    float sm = 0.0f;
    for (int i = tid; i < B; i += 256) sm += __expf(s[i] - mx);
    red[tid] = sm; __syncthreads();
    for (int st = 128; st > 0; st >>= 1) {
        if (tid < st) red[tid] += red[tid + st];
        __syncthreads();
    }
    const float inv = 1.0f / red[0];
    for (int i = tid; i < B; i += 256) att[(size_t)m * B + i] = __expf(s[i] - mx) * inv;
}

// ================= kernel 5: pa = sum_m pl*att + 1 =================
__global__ __launch_bounds__(256) void k_pa(const float* __restrict__ pl,
                                            const float* __restrict__ att,
                                            float* __restrict__ pa) {
    int idx = blockIdx.x * 256 + threadIdx.x;   // B*L total
    int b = idx >> 7, l = idx & 127;
    float v = 1.0f;
#pragma unroll
    for (int m = 0; m < MP; ++m)
        v = fmaf(pl[((size_t)m * B + b) * LDIM + l], att[(size_t)m * B + b], v);
    pa[idx] = v;
}

// ================= kernel 6: user/item attention + final projection + sigmoid ==========
// 1 WG = 16 b rows; 256 threads; LDS staged with CDNA5 async global->LDS copies.
__global__ __launch_bounds__(256) void k_final(const float* __restrict__ ul,
                                               const float* __restrict__ il,
                                               const float* __restrict__ pa,
                                               const float* __restrict__ Wua,
                                               const float* __restrict__ bua,
                                               const float* __restrict__ Wia,
                                               const float* __restrict__ bia,
                                               const float* __restrict__ Wp,
                                               const float* __restrict__ bp,
                                               float* __restrict__ out) {
    __shared__ __align__(16) float sUL[16 * 128], sIL[16 * 128], sPA[16 * 128];
    __shared__ __align__(16) float sUA[16 * 128], sIA[16 * 128];
    const int b0 = blockIdx.x * 16, tid = threadIdx.x;

    // async global->LDS staging: 3 x 8 KB pure copies, no conversion -> ASYNCcnt path
    {
        const unsigned ldsUL = (unsigned)(uintptr_t)(void*)sUL;
        const unsigned ldsIL = (unsigned)(uintptr_t)(void*)sIL;
        const unsigned ldsPA = (unsigned)(uintptr_t)(void*)sPA;
        const float* gUL = ul + (size_t)b0 * 128;
        const float* gIL = il + (size_t)b0 * 128;
        const float* gPA = pa + (size_t)b0 * 128;
#pragma unroll
        for (int i = 0; i < 2; ++i) {
            int boff = (tid + i * 256) * 16;   // 512 x b128 per array
            async_load_b128(ldsUL + boff, gUL, boff);
            async_load_b128(ldsIL + boff, gIL, boff);
            async_load_b128(ldsPA + boff, gPA, boff);
        }
        wait_async0();
    }
    __syncthreads();

    // ---- user attention: g = relu([ul|pa] @ Wua + bua); ua = ul * softmax_L(g)
    for (int pass = 0; pass < 8; ++pass) {
        int bi = (tid >> 7) + pass * 2, n = tid & 127;
        float a = bua[n];
        const float* r1 = sUL + bi * 128;
        const float* r2 = sPA + bi * 128;
        for (int k = 0; k < 128; ++k) a = fmaf(r1[k], Wua[k * 128 + n], a);
        for (int k = 0; k < 128; ++k) a = fmaf(r2[k], Wua[(128 + k) * 128 + n], a);
        sUA[bi * 128 + n] = fmaxf(a, 0.0f);
    }
    __syncthreads();
    float vals[8];
    for (int pass = 0; pass < 8; ++pass) {
        int bi = (tid >> 7) + pass * 2, n = tid & 127;
        const float* g = sUA + bi * 128;
        float mx = -3.4e38f;
        for (int k = 0; k < 128; ++k) mx = fmaxf(mx, g[k]);
        float sm = 0.0f;
        for (int k = 0; k < 128; ++k) sm += __expf(g[k] - mx);
        vals[pass] = sUL[bi * 128 + n] * __expf(g[n] - mx) / sm;
    }
    __syncthreads();
    for (int pass = 0; pass < 8; ++pass) {
        int bi = (tid >> 7) + pass * 2, n = tid & 127;
        sUA[bi * 128 + n] = vals[pass];
    }
    __syncthreads();

    // ---- item attention: same with il / Wia / bia
    for (int pass = 0; pass < 8; ++pass) {
        int bi = (tid >> 7) + pass * 2, n = tid & 127;
        float a = bia[n];
        const float* r1 = sIL + bi * 128;
        const float* r2 = sPA + bi * 128;
        for (int k = 0; k < 128; ++k) a = fmaf(r1[k], Wia[k * 128 + n], a);
        for (int k = 0; k < 128; ++k) a = fmaf(r2[k], Wia[(128 + k) * 128 + n], a);
        sIA[bi * 128 + n] = fmaxf(a, 0.0f);
    }
    __syncthreads();
    for (int pass = 0; pass < 8; ++pass) {
        int bi = (tid >> 7) + pass * 2, n = tid & 127;
        const float* g = sIA + bi * 128;
        float mx = -3.4e38f;
        for (int k = 0; k < 128; ++k) mx = fmaxf(mx, g[k]);
        float sm = 0.0f;
        for (int k = 0; k < 128; ++k) sm += __expf(g[k] - mx);
        vals[pass] = sIL[bi * 128 + n] * __expf(g[n] - mx) / sm;
    }
    __syncthreads();
    for (int pass = 0; pass < 8; ++pass) {
        int bi = (tid >> 7) + pass * 2, n = tid & 127;
        sIA[bi * 128 + n] = vals[pass];
    }
    __syncthreads();

    // ---- out = sigmoid([ua|pa|ia] @ Wp + bp)
    if (tid < 16) {
        float a = bp[0];
        for (int n = 0; n < 128; ++n) {
            a = fmaf(sUA[tid * 128 + n], Wp[n], a);
            a = fmaf(sPA[tid * 128 + n], Wp[128 + n], a);
            a = fmaf(sIA[tid * 128 + n], Wp[256 + n], a);
        }
        out[b0 + tid] = 1.0f / (1.0f + __expf(-a));
    }
}

// ================================================================================
extern "C" void kernel_launch(void* const* d_in, const int* in_sizes, int n_in,
                              void* d_out, int out_size, void* d_ws, size_t ws_size,
                              hipStream_t stream) {
    (void)in_sizes; (void)n_in; (void)out_size; (void)ws_size;

    const int*   ui    = (const int*)d_in[0];
    const int*   ii    = (const int*)d_in[1];
    const float* path  = (const float*)d_in[2];
    const float* uemb  = (const float*)d_in[3];
    const float* iemb  = (const float*)d_in[4];
    const float* Wconv = (const float*)d_in[5];
    const float* bconv = (const float*)d_in[6];
    const float* W1    = (const float*)d_in[7];
    const float* b1    = (const float*)d_in[8];
    const float* W2    = (const float*)d_in[9];
    const float* b2    = (const float*)d_in[10];
    const float* Wua   = (const float*)d_in[11];
    const float* bua   = (const float*)d_in[12];
    const float* Wia   = (const float*)d_in[13];
    const float* bia   = (const float*)d_in[14];
    const float* Wp    = (const float*)d_in[15];
    const float* bp    = (const float*)d_in[16];
    float* out = (float*)d_out;

    float* ws = (float*)d_ws;
    float* pl = ws;                                  // [M,B,L]
    float* ul = pl + (size_t)MP * B * LDIM;          // [B,L]
    float* il = ul + (size_t)B * LDIM;               // [B,L]
    float* sc = il + (size_t)B * LDIM;               // [M,B]
    float* at = sc + (size_t)MP * B;                 // [M,B]
    float* pa = at + (size_t)MP * B;                 // [B,L]

    k_gather   <<<B, 128, 0, stream>>>(ui, ii, uemb, iemb, ul, il);
    k_conv_pool<<<MP * (B / 4), 256, 0, stream>>>(path, Wconv, bconv, pl);
    k_score    <<<MP * (B / 64), 256, 0, stream>>>(ul, il, pl, W1, b1, W2, b2, sc);
    k_softmax_b<<<MP, 256, 0, stream>>>(sc, at);
    k_pa       <<<(B * LDIM) / 256, 256, 0, stream>>>(pl, at, pa);
    k_final    <<<B / 16, 256, 0, stream>>>(ul, il, pa, Wua, bua, Wia, bia, Wp, bp, out);
}
